// Net_72232759984655
// MI455X (gfx1250) — compile-verified, matrix operand order checked by reading, not run on hardware
//
#include <hip/hip_runtime.h>

#define B_   8
#define N_   2048
#define KNN  16
#define PTS  (B_*N_)

typedef __attribute__((ext_vector_type(16))) _Float16 v16h;
typedef __attribute__((ext_vector_type(8)))  _Float16 v8h;
typedef __attribute__((ext_vector_type(8)))  float    v8f;
typedef __attribute__((ext_vector_type(2)))  float    v2f;

__device__ __forceinline__ float lrelu_f(float v){ return v > 0.f ? v : 0.01f*v; }

// ---------- weight convert + transpose + zero-pad: dst[n*CinPad+k] = (k<Cin)? w[k*Cout+n] : 0
__global__ void wt_cvt(const float* __restrict__ w, _Float16* __restrict__ dst,
                       int Cin, int Cout, int CinPad){
    int t = blockIdx.x*blockDim.x + threadIdx.x;
    if (t >= Cout*CinPad) return;
    int n = t / CinPad, k = t - n*CinPad;
    dst[t] = (k < Cin) ? (_Float16)w[k*Cout + n] : (_Float16)0.f;
}

// ---------- per-point squared norms
__global__ void rowsq(const float* __restrict__ feat, float* __restrict__ sq, int C){
    int t = blockIdx.x*blockDim.x + threadIdx.x;
    if (t >= PTS) return;
    const float* f = feat + (size_t)t*C;
    float s = 0.f;
    for (int c = 0; c < C; ++c) s += f[c]*f[c];
    sq[t] = s;
}

// C==3 special fragment load (no OOB, half-wave select: lanes>=16 hold K={2,pad0})
__device__ __forceinline__ v2f ldpair3(const float* __restrict__ p, int hi){
    v2f r;
    if (hi == 0){ r.x = p[0]; r.y = p[1]; }
    else        { r.x = p[2]; r.y = 0.f;  }
    return r;
}

// ---------- fp32 Gram strip (16 rows x 64 cols) via V_WMMA_F32_16X16X4_F32
// epilogue: d2 = sqi + sqj - 2G.  C is compile-time -> unconditional b64 loads, full unroll.
template<int C>
__global__ __launch_bounds__(32) void dist_wmma(const float* __restrict__ feat,
        const float* __restrict__ sq, float* __restrict__ d2, int b){
    int tile = blockIdx.x;                  // (N/16) * (N/64) = 128*32 strips
    int ti = tile >> 5, tj4 = tile & 31;
    int i0 = ti*16, j0 = tj4*64;
    int lane = threadIdx.x;
    int l15 = lane & 15, hi = lane >> 4;
    const float* fb = feat + (size_t)b*N_*C;
    const float* ar  = fb + (size_t)(i0 + l15)*C;
    const float* br0 = fb + (size_t)(j0      + l15)*C;
    const float* br1 = fb + (size_t)(j0 + 16 + l15)*C;
    const float* br2 = fb + (size_t)(j0 + 32 + l15)*C;
    const float* br3 = fb + (size_t)(j0 + 48 + l15)*C;
    v8f acc0 = {}, acc1 = {}, acc2 = {}, acc3 = {};
    if constexpr (C == 3){
        v2f a  = ldpair3(ar,  hi);
        v2f b0 = ldpair3(br0, hi);
        v2f b1 = ldpair3(br1, hi);
        v2f b2 = ldpair3(br2, hi);
        v2f b3 = ldpair3(br3, hi);
        acc0 = __builtin_amdgcn_wmma_f32_16x16x4_f32(false, a, false, b0, (short)0, acc0, false, false);
        acc1 = __builtin_amdgcn_wmma_f32_16x16x4_f32(false, a, false, b1, (short)0, acc1, false, false);
        acc2 = __builtin_amdgcn_wmma_f32_16x16x4_f32(false, a, false, b2, (short)0, acc2, false, false);
        acc3 = __builtin_amdgcn_wmma_f32_16x16x4_f32(false, a, false, b3, (short)0, acc3, false, false);
    } else {
        #pragma unroll
        for (int k0 = 0; k0 < C; k0 += 4){
            // 32-bit A 16x4 layout: lanes<16 hold K={k0,k0+1}, lanes>=16 hold K={k0+2,k0+3}
            int ka = k0 + 2*hi;
            v2f a  = *(const v2f*)(ar  + ka);
            v2f b0 = *(const v2f*)(br0 + ka);
            v2f b1 = *(const v2f*)(br1 + ka);
            v2f b2 = *(const v2f*)(br2 + ka);
            v2f b3 = *(const v2f*)(br3 + ka);
            acc0 = __builtin_amdgcn_wmma_f32_16x16x4_f32(false, a, false, b0, (short)0, acc0, false, false);
            acc1 = __builtin_amdgcn_wmma_f32_16x16x4_f32(false, a, false, b1, (short)0, acc1, false, false);
            acc2 = __builtin_amdgcn_wmma_f32_16x16x4_f32(false, a, false, b2, (short)0, acc2, false, false);
            acc3 = __builtin_amdgcn_wmma_f32_16x16x4_f32(false, a, false, b3, (short)0, acc3, false, false);
        }
    }
    const float* sqb = sq + b*N_;
    float sqj0 = sqb[j0      + l15];
    float sqj1 = sqb[j0 + 16 + l15];
    float sqj2 = sqb[j0 + 32 + l15];
    float sqj3 = sqb[j0 + 48 + l15];
    #pragma unroll
    for (int r = 0; r < 8; ++r){
        int i = i0 + 8*hi + r;               // C/D layout: M = r + 8*(lane>=16), N = lane&15
        float sqi = sqb[i];
        float* dro = d2 + (size_t)i*N_ + l15;
        dro[j0     ] = sqi + sqj0 - 2.f*acc0[r];
        dro[j0 + 16] = sqi + sqj1 - 2.f*acc1[r];
        dro[j0 + 32] = sqi + sqj2 - 2.f*acc2[r];
        dro[j0 + 48] = sqi + sqj3 - 2.f*acc3[r];
    }
}

// ---------- top-16 smallest per row, register-resident insertion list
__global__ void topk_kernel(const float* __restrict__ d2, int* __restrict__ idx, int b){
    int row = blockIdx.x*blockDim.x + threadIdx.x;
    if (row >= N_) return;
    const float* dr = d2 + (size_t)row*N_;
    float best[KNN]; int bi[KNN];
    #pragma unroll
    for (int i = 0; i < KNN; ++i){ best[i] = 3.4e38f; bi[i] = 0; }
    for (int m = 0; m < N_; ++m){
        float d = dr[m];
        if (d < best[KNN-1]){
            best[KNN-1] = d; bi[KNN-1] = m;
            #pragma unroll
            for (int i = KNN-1; i > 0; --i){
                if (best[i] < best[i-1]){
                    float tf = best[i-1]; best[i-1] = best[i]; best[i] = tf;
                    int   tn = bi[i-1];   bi[i-1]   = bi[i];   bi[i]   = tn;
                }
            }
        }
    }
    int* op = idx + ((size_t)b*N_ + row)*KNN;
    #pragma unroll
    for (int i = 0; i < KNN; ++i) op[i] = bi[i];
}

// ---------- one MLP layer: [16 x CinPad](LDS,f16) x WT[Cout x CinPad](global,f16) -> LDS f16
// A-frag (16-bit 16x32): lane<16 -> row=lane, K = k0..k0+7 & k0+16..k0+23; lane>=16 -> +8
template<int CinPad, int Cout, int OStride>
__device__ __forceinline__ void wmma_layer(const _Float16* __restrict__ A,
        const _Float16* __restrict__ Wt, const float* __restrict__ bias,
        _Float16* __restrict__ O){
    int lane = threadIdx.x & 31;
    int l15  = lane & 15;
    int kb   = (lane >> 4) << 3;
    int mb   = (lane >> 4) << 3;
    #pragma unroll
    for (int n0 = 0; n0 < Cout; n0 += 16){
        v8f acc = {};
        int col = n0 + l15;
        #pragma unroll
        for (int k0 = 0; k0 < CinPad; k0 += 32){
            const _Float16* ap = A + l15*CinPad + k0 + kb;
            v8h alo = *(const v8h*)ap;
            v8h ahi = *(const v8h*)(ap + 16);
            v16h av = __builtin_shufflevector(alo, ahi, 0,1,2,3,4,5,6,7,8,9,10,11,12,13,14,15);
            const _Float16* bp = Wt + (size_t)col*CinPad + k0 + kb;
            v8h blo = *(const v8h*)bp;
            v8h bhi = *(const v8h*)(bp + 16);
            v16h bv = __builtin_shufflevector(blo, bhi, 0,1,2,3,4,5,6,7,8,9,10,11,12,13,14,15);
            acc = __builtin_amdgcn_wmma_f32_16x16x32_f16(false, av, false, bv,
                                                         (short)0, acc, false, false);
        }
        float bs = bias[col];
        #pragma unroll
        for (int r = 0; r < 8; ++r)
            O[(mb + r)*OStride + col] = (_Float16)lrelu_f(acc[r] + bs);
    }
}

// ---------- final layer: bias + lrelu + sum over the 16 neighbor rows (M dim) -> global f32
template<int CinPad, int Cout>
__device__ __forceinline__ void wmma_layer_sum(const _Float16* __restrict__ A,
        const _Float16* __restrict__ Wt, const float* __restrict__ bias,
        float* __restrict__ outp){
    int lane = threadIdx.x & 31;
    int l15  = lane & 15;
    int kb   = (lane >> 4) << 3;
    #pragma unroll
    for (int n0 = 0; n0 < Cout; n0 += 16){
        v8f acc = {};
        int col = n0 + l15;
        #pragma unroll
        for (int k0 = 0; k0 < CinPad; k0 += 32){
            const _Float16* ap = A + l15*CinPad + k0 + kb;
            v8h alo = *(const v8h*)ap;
            v8h ahi = *(const v8h*)(ap + 16);
            v16h av = __builtin_shufflevector(alo, ahi, 0,1,2,3,4,5,6,7,8,9,10,11,12,13,14,15);
            const _Float16* bp = Wt + (size_t)col*CinPad + k0 + kb;
            v8h blo = *(const v8h*)bp;
            v8h bhi = *(const v8h*)(bp + 16);
            v16h bv = __builtin_shufflevector(blo, bhi, 0,1,2,3,4,5,6,7,8,9,10,11,12,13,14,15);
            acc = __builtin_amdgcn_wmma_f32_16x16x32_f16(false, av, false, bv,
                                                         (short)0, acc, false, false);
        }
        float bs = bias[col];
        float part = 0.f;
        #pragma unroll
        for (int r = 0; r < 8; ++r) part += lrelu_f(acc[r] + bs);
        part += __shfl_xor(part, 16, 32);      // combine M=0..7 with M=8..15 halves
        if (lane < 16) outp[col] = part;
    }
}

// ---------- stage 1: one wave per point; A0=[16][6->pad32]; MLP 6->16->64->32; sum K
__global__ __launch_bounds__(32) void mlp_stage1(const float* __restrict__ x,
        const int* __restrict__ idx,
        const _Float16* __restrict__ w1t, const float* __restrict__ b1,
        const _Float16* __restrict__ w2t, const float* __restrict__ b2,
        const _Float16* __restrict__ w3t, const float* __restrict__ b3,
        float* __restrict__ out){
    __shared__ __align__(16) _Float16 bufA[16*64];
    __shared__ __align__(16) _Float16 bufB[16*32];
    int pt = blockIdx.x;
    int b = pt >> 11, n = pt & (N_-1);
    int lane = threadIdx.x, l15 = lane & 15, hs = lane >> 4;
    int nbr = idx[pt*KNN + l15];
    const float* fc = x + ((size_t)b*N_ + n)*3;
    const float* fn = x + ((size_t)b*N_ + nbr)*3;
    #pragma unroll
    for (int c = 0; c < 16; ++c){
        int cc = hs*16 + c;
        float v = 0.f;
        if (cc < 3) v = fc[cc]; else if (cc < 6) v = fn[cc-3];
        bufA[l15*32 + cc] = (_Float16)v;
    }
    #pragma unroll
    for (int c = 0; c < 16; ++c) bufB[lane*16 + c] = (_Float16)0.f;  // pad cols 16..31
    __syncthreads();
    wmma_layer<32,16,32>(bufA, w1t, b1, bufB);
    __syncthreads();
    wmma_layer<32,64,64>(bufB, w2t, b2, bufA);
    __syncthreads();
    wmma_layer_sum<64,32>(bufA, w3t, b3, out + (size_t)pt*32);
}

// ---------- stage 2: A0=[16][64]; MLP 64->256->128; sum K
__global__ __launch_bounds__(32) void mlp_stage2(const float* __restrict__ x2,
        const int* __restrict__ idx,
        const _Float16* __restrict__ w1t, const float* __restrict__ b1,
        const _Float16* __restrict__ w2t, const float* __restrict__ b2,
        float* __restrict__ out){
    __shared__ __align__(16) _Float16 bufA[16*64];
    __shared__ __align__(16) _Float16 bufH[16*256];
    int pt = blockIdx.x;
    int b = pt >> 11, n = pt & (N_-1);
    int lane = threadIdx.x, l15 = lane & 15, hs = lane >> 4;
    int nbr = idx[pt*KNN + l15];
    const float* f = x2 + ((size_t)b*N_ + (hs ? nbr : n))*32;
    #pragma unroll
    for (int c = 0; c < 32; ++c) bufA[l15*64 + hs*32 + c] = (_Float16)f[c];
    __syncthreads();
    wmma_layer<64,256,256>(bufA, w1t, b1, bufH);
    __syncthreads();
    wmma_layer_sum<256,128>(bufH, w2t, b2, out + (size_t)pt*128);
}

// ---------- stage 3: A0=[16][256]; MLP 256->128; sum K
__global__ __launch_bounds__(32) void mlp_stage3(const float* __restrict__ x3,
        const int* __restrict__ idx,
        const _Float16* __restrict__ wt, const float* __restrict__ bias,
        float* __restrict__ out){
    __shared__ __align__(16) _Float16 bufA[16*256];
    int pt = blockIdx.x;
    int b = pt >> 11, n = pt & (N_-1);
    int lane = threadIdx.x, l15 = lane & 15, hs = lane >> 4;
    int nbr = idx[pt*KNN + l15];
    const float* f = x3 + ((size_t)b*N_ + (hs ? nbr : n))*128;
    #pragma unroll 8
    for (int c = 0; c < 128; ++c) bufA[l15*256 + hs*128 + c] = (_Float16)f[c];
    __syncthreads();
    wmma_layer_sum<256,128>(bufA, wt, bias, out + (size_t)pt*128);
}

// ---------- global max-pool over points
__global__ void maxpool_kernel(const float* __restrict__ x4, float* __restrict__ x5){
    int b = blockIdx.x, c = threadIdx.x;
    float m = -3.4e38f;
    for (int n = 0; n < N_; ++n)
        m = fmaxf(m, x4[((size_t)b*N_ + n)*128 + c]);
    x5[b*128 + c] = m;
}

// ---------- tiny FC head (fp32 scalar; negligible FLOPs)
__global__ void fc_head(const float* __restrict__ x5,
        const float* __restrict__ w1, const float* __restrict__ b1,
        const float* __restrict__ w2, const float* __restrict__ b2,
        const float* __restrict__ w3, const float* __restrict__ b3,
        float* __restrict__ out){
    __shared__ float s0[128], s1[128];
    int b = blockIdx.x, t = threadIdx.x;
    s0[t] = x5[b*128 + t];
    __syncthreads();
    float a = b1[t];
    for (int k = 0; k < 128; ++k) a += s0[k]*w1[k*128 + t];
    s1[t] = lrelu_f(a);
    __syncthreads();
    a = b2[t];
    for (int k = 0; k < 128; ++k) a += s1[k]*w2[k*128 + t];
    __syncthreads();
    s0[t] = lrelu_f(a);
    __syncthreads();
    float* ob = out + (size_t)b*6144;
    for (int j = t; j < 6144; j += 128){
        float acc = b3[j];
        for (int k = 0; k < 128; ++k) acc += s0[k]*w3[(size_t)k*6144 + j];
        ob[j] = acc;
    }
}

extern "C" void kernel_launch(void* const* d_in, const int* in_sizes, int n_in,
                              void* d_out, int out_size, void* d_ws, size_t ws_size,
                              hipStream_t stream){
    const float* x    = (const float*)d_in[0];
    const float* h1w1 = (const float*)d_in[1];  const float* h1b1 = (const float*)d_in[2];
    const float* h1w2 = (const float*)d_in[3];  const float* h1b2 = (const float*)d_in[4];
    const float* h1w3 = (const float*)d_in[5];  const float* h1b3 = (const float*)d_in[6];
    const float* h2w1 = (const float*)d_in[7];  const float* h2b1 = (const float*)d_in[8];
    const float* h2w2 = (const float*)d_in[9];  const float* h2b2 = (const float*)d_in[10];
    const float* h3w1 = (const float*)d_in[11]; const float* h3b1 = (const float*)d_in[12];
    const float* fc1w = (const float*)d_in[13]; const float* fc1b = (const float*)d_in[14];
    const float* fc2w = (const float*)d_in[15]; const float* fc2b = (const float*)d_in[16];
    const float* fc3w = (const float*)d_in[17]; const float* fc3b = (const float*)d_in[18];
    (void)in_sizes; (void)n_in; (void)out_size; (void)ws_size;

    char* ws = (char*)d_ws;
    size_t off = 0;
    auto walloc = [&](size_t bytes)->char*{
        char* p = ws + off;
        off = (off + bytes + 255) & ~(size_t)255;
        return p;
    };
    _Float16* w11t = (_Float16*)walloc(16*32*2);
    _Float16* w12t = (_Float16*)walloc(64*32*2);
    _Float16* w13t = (_Float16*)walloc(32*64*2);
    _Float16* w21t = (_Float16*)walloc(256*64*2);
    _Float16* w22t = (_Float16*)walloc(128*256*2);
    _Float16* w31t = (_Float16*)walloc(128*256*2);
    float* sq  = (float*)walloc((size_t)PTS*4);
    float* d2  = (float*)walloc((size_t)N_*N_*4);          // reused per batch (stream-ordered)
    int*   idx = (int*)  walloc((size_t)PTS*KNN*4);        // reused per stage
    float* x2  = (float*)walloc((size_t)PTS*32*4);
    float* x3  = (float*)walloc((size_t)PTS*128*4);
    float* x4  = (float*)walloc((size_t)PTS*128*4);
    float* x5  = (float*)walloc((size_t)B_*128*4);

    auto cvt = [&](const float* w, _Float16* dst, int Cin, int Cout, int CinPad){
        int total = Cout*CinPad;
        wt_cvt<<<(total+255)/256, 256, 0, stream>>>(w, dst, Cin, Cout, CinPad);
    };
    cvt(h1w1, w11t,   6,  16,  32);
    cvt(h1w2, w12t,  16,  64,  32);
    cvt(h1w3, w13t,  64,  32,  64);
    cvt(h2w1, w21t,  64, 256,  64);
    cvt(h2w2, w22t, 256, 128, 256);
    cvt(h3w1, w31t, 256, 128, 256);

    const int DGRID = (N_/16)*(N_/64);   // 16x64 strips per batch

    // ---- stage 1 KNN (C=3) + MLP
    rowsq<<<(PTS+255)/256, 256, 0, stream>>>(x, sq, 3);
    for (int b = 0; b < B_; ++b){
        dist_wmma<3><<<DGRID, 32, 0, stream>>>(x, sq, d2, b);
        topk_kernel<<<N_/128, 128, 0, stream>>>(d2, idx, b);
    }
    mlp_stage1<<<PTS, 32, 0, stream>>>(x,  idx, w11t, h1b1, w12t, h1b2, w13t, h1b3, x2);

    // ---- stage 2 KNN (C=32) + MLP
    rowsq<<<(PTS+255)/256, 256, 0, stream>>>(x2, sq, 32);
    for (int b = 0; b < B_; ++b){
        dist_wmma<32><<<DGRID, 32, 0, stream>>>(x2, sq, d2, b);
        topk_kernel<<<N_/128, 128, 0, stream>>>(d2, idx, b);
    }
    mlp_stage2<<<PTS, 32, 0, stream>>>(x2, idx, w21t, h2b1, w22t, h2b2, x3);

    // ---- stage 3 KNN (C=128) + MLP
    rowsq<<<(PTS+255)/256, 256, 0, stream>>>(x3, sq, 128);
    for (int b = 0; b < B_; ++b){
        dist_wmma<128><<<DGRID, 32, 0, stream>>>(x3, sq, d2, b);
        topk_kernel<<<N_/128, 128, 0, stream>>>(d2, idx, b);
    }
    mlp_stage3<<<PTS, 32, 0, stream>>>(x3, idx, w31t, h3b1, x4);

    maxpool_kernel<<<B_, 128, 0, stream>>>(x4, x5);
    fc_head<<<B_, 128, 0, stream>>>(x5, fc1w, fc1b, fc2w, fc2b, fc3w, fc3b, (float*)d_out);
}
